// GNN_45689862095097
// MI455X (gfx1250) — compile-verified, matrix operand order checked by reading, not run on hardware
//
#include <hip/hip_runtime.h>
#include <hip/hip_bf16.h>

#define NN    50000
#define EE    800000
#define FNODE 64
#define FEDGE 16
#define HIDD  32
#define HEADS 4
#define HC    128
#define GGR   2000
#define ETOT  (EE + NN)

typedef __attribute__((ext_vector_type(16))) _Float16 v16h;
typedef __attribute__((ext_vector_type(8)))  float    v8f;

// ---- ordered-uint encoding for float atomic max ----
__device__ __forceinline__ unsigned f2ord(float f) {
  unsigned u = __float_as_uint(f);
  return (u & 0x80000000u) ? ~u : (u | 0x80000000u);
}
__device__ __forceinline__ float ord2f(unsigned u) {
  return (u & 0x80000000u) ? __uint_as_float(u & 0x7FFFFFFFu) : __uint_as_float(~u);
}

// =====================================================================
// Row-major WMMA GEMM: C[M x Nc] = A[M x K] @ B[K x Nc] (+bias, +relu).
// M % 16 == 0, K % 32 == 0, Nc % 32 == 0.
// Each wave owns a 16x32 output slice: one A fragment feeds TWO
// v_wmma_f32_16x16x32_f16 issues per K-step (A reuse, back-to-back XDL).
// 4 waves/block cover 128 output columns. f16 inputs, f32 accumulate.
// =====================================================================
__global__ void wmma_gemm(const float* __restrict__ A, const float* __restrict__ B,
                          const float* __restrict__ bias, float* __restrict__ C,
                          int M, int K, int Nc, int fuse_relu) {
  const int wave = threadIdx.x >> 5;
  const int lane = threadIdx.x & 31;
  const int hh   = (lane >> 4) & 1;     // lane half (0: lanes 0-15, 1: 16-31)
  const int l16  = lane & 15;
  const int rowT = blockIdx.x;
  const int colP = blockIdx.y * 4 + wave;          // 32-column slice index
  if (colP * 32 >= Nc) return;                     // wave-uniform: EXEC all-1s
  const int row  = rowT * 16 + l16;                // A row for this lane
  const int col0 = colP * 32 + l16;                // first 16-col tile
  const int col1 = col0 + 16;                      // second 16-col tile

  v8f acc0 = {}, acc1 = {};
  for (int k0 = 0; k0 < K; k0 += 32) {
    v16h a, b0, b1;
#pragma unroll
    for (int j = 0; j < 8; ++j) {
      // 16-bit A 16x32 layout: VGPR j<4 -> K=2j+8*half; j>=4 -> K=16+2(j-4)+8*half
      const int kb = k0 + ((j < 4) ? (2 * j) : (16 + 2 * (j - 4))) + 8 * hh;
      const float2 av = *(const float2*)(A + (size_t)row * K + kb);  // kb even, 8B aligned
      a[2 * j]     = (_Float16)av.x;
      a[2 * j + 1] = (_Float16)av.y;
      const float* Brow0 = B + (size_t)kb * Nc;
      const float* Brow1 = B + (size_t)(kb + 1) * Nc;
      b0[2 * j]     = (_Float16)Brow0[col0];
      b0[2 * j + 1] = (_Float16)Brow1[col0];
      b1[2 * j]     = (_Float16)Brow0[col1];
      b1[2 * j + 1] = (_Float16)Brow1[col1];
    }
    acc0 = __builtin_amdgcn_wmma_f32_16x16x32_f16(false, a, false, b0, (short)0, acc0,
                                                  false, false);
    acc1 = __builtin_amdgcn_wmma_f32_16x16x32_f16(false, a, false, b1, (short)0, acc1,
                                                  false, false);
  }

  // Branch-free epilogue: bias loaded once per lane, relu flag hoisted.
  const float bv0 = bias ? bias[col0] : 0.f;
  const float bv1 = bias ? bias[col1] : 0.f;
#pragma unroll
  for (int r = 0; r < 8; ++r) {
    const int orow = rowT * 16 + r + 8 * hh;       // C VGPR r -> row r (+8 upper lanes)
    float v0 = acc0[r] + bv0;
    float v1 = acc1[r] + bv1;
    if (fuse_relu) { v0 = fmaxf(v0, 0.f); v1 = fmaxf(v1, 0.f); }
    C[(size_t)orow * Nc + col0] = v0;
    C[(size_t)orow * Nc + col1] = v1;
  }
}

// ---- self-loop attr: per-dst count and edge-feature sum ----
__global__ void k_count_loop(const int* __restrict__ dst0, const float* __restrict__ ef,
                             float* __restrict__ cnt, float* __restrict__ loop_sum) {
  int e = blockIdx.x * blockDim.x + threadIdx.x;
  if (e >= EE) return;
  int d = dst0[e];
  atomicAdd(&cnt[d], 1.0f);
#pragma unroll
  for (int f = 0; f < FEDGE; ++f)
    atomicAdd(&loop_sum[(size_t)d * FEDGE + f], ef[(size_t)e * FEDGE + f]);
}

__global__ void k_loop_div(float* __restrict__ loop_attr, const float* __restrict__ cnt) {
  int i = blockIdx.x * blockDim.x + threadIdx.x;
  if (i >= NN * FEDGE) return;
  float c = cnt[i / FEDGE];
  c = c > 1.f ? c : 1.f;
  loop_attr[i] /= c;
}

// ---- per-node attention scalars: al_s, al_d ----
__global__ void k_node_att(const float* __restrict__ xs, const float* __restrict__ a_src,
                           const float* __restrict__ a_dst, float* __restrict__ al_s,
                           float* __restrict__ al_d) {
  int i = blockIdx.x * blockDim.x + threadIdx.x;   // n*HEADS + h
  if (i >= NN * HEADS) return;
  int n = i / HEADS, h = i % HEADS;
  const float* xp = xs + (size_t)n * HC + h * HIDD;
  float ss = 0.f, sd = 0.f;
#pragma unroll 8
  for (int c = 0; c < HIDD; ++c) {
    float v = xp[c];
    ss += v * a_src[h * HIDD + c];
    sd += v * a_dst[h * HIDD + c];
  }
  al_s[i] = ss;
  al_d[i] = sd;
}

// ---- fold We and a_edge: w_att[f,h] = sum_c We[f, h*C+c] * a_edge[h,c] ----
__global__ void k_watt(const float* __restrict__ We, const float* __restrict__ a_edge,
                       float* __restrict__ w_att) {
  int i = threadIdx.x;
  if (i >= FEDGE * HEADS) return;
  int f = i / HEADS, h = i % HEADS;
  float s = 0.f;
  for (int c = 0; c < HIDD; ++c) s += We[f * HC + h * HIDD + c] * a_edge[h * HIDD + c];
  w_att[i] = s;
}

__global__ void k_init_m(unsigned* __restrict__ m, int n) {
  int i = blockIdx.x * blockDim.x + threadIdx.x;
  if (i < n) m[i] = 0x007FFFFFu;   // f2ord(-inf)
}

// ---- edge logits + segment max (ordered-uint atomicMax) ----
__global__ void k_logits(const int* __restrict__ src0, const int* __restrict__ dst0,
                         const float* __restrict__ ef, const float* __restrict__ loop_attr,
                         const float* __restrict__ al_s, const float* __restrict__ al_d,
                         const float* __restrict__ w_att, float* __restrict__ logits,
                         unsigned* __restrict__ mord) {
  int e = blockIdx.x * blockDim.x + threadIdx.x;
  if (e >= ETOT) return;
  int s, d;
  const float* eap;
  if (e < EE) { s = src0[e]; d = dst0[e]; eap = ef + (size_t)e * FEDGE; }
  else        { s = d = e - EE;           eap = loop_attr + (size_t)(e - EE) * FEDGE; }
  float ale[HEADS] = {0.f, 0.f, 0.f, 0.f};
#pragma unroll
  for (int f = 0; f < FEDGE; ++f) {
    float v = eap[f];
#pragma unroll
    for (int h = 0; h < HEADS; ++h) ale[h] += v * w_att[f * HEADS + h];
  }
#pragma unroll
  for (int h = 0; h < HEADS; ++h) {
    float lg = al_s[s * HEADS + h] + al_d[d * HEADS + h] + ale[h];
    lg = lg > 0.f ? lg : 0.2f * lg;                       // leaky_relu(., 0.2)
    logits[(size_t)e * HEADS + h] = lg;
    atomicMax(&mord[d * HEADS + h], f2ord(lg));
  }
}

// ---- p = exp(logit - max), denom = segment_sum(p) ----
__global__ void k_expsum(const int* __restrict__ dst0, float* __restrict__ logits,
                         const unsigned* __restrict__ mord, float* __restrict__ denom) {
  int e = blockIdx.x * blockDim.x + threadIdx.x;
  if (e >= ETOT) return;
  int d = (e < EE) ? dst0[e] : (e - EE);
#pragma unroll
  for (int h = 0; h < HEADS; ++h) {
    float mm = ord2f(mord[d * HEADS + h]);
    float p = __expf(logits[(size_t)e * HEADS + h] - mm);
    logits[(size_t)e * HEADS + h] = p;
    atomicAdd(&denom[d * HEADS + h], p);
  }
}

// ---- out[dst] += alpha * xs[src], one thread per (edge, channel) ----
__global__ void k_aggregate(const int* __restrict__ src0, const int* __restrict__ dst0,
                            const float* __restrict__ p, const float* __restrict__ denom,
                            const float* __restrict__ xs, float* __restrict__ out) {
  size_t idx = (size_t)blockIdx.x * blockDim.x + threadIdx.x;
  if (idx >= (size_t)ETOT * HC) return;
  int e = (int)(idx / HC);
  int ch = (int)(idx % HC);
  int h = ch / HIDD;
  int s, d;
  if (e < EE) { s = src0[e]; d = dst0[e]; } else { s = d = e - EE; }
  float alpha = p[(size_t)e * HEADS + h] / (denom[d * HEADS + h] + 1e-16f);
  atomicAdd(&out[(size_t)d * HC + ch], alpha * xs[(size_t)s * HC + ch]);
}

__global__ void k_bias_relu(const float* __restrict__ acc, const float* __restrict__ b,
                            float* __restrict__ hnext) {
  int i = blockIdx.x * blockDim.x + threadIdx.x;
  if (i >= NN * HC) return;
  float v = acc[i] + b[i % HC];
  hnext[i] = v > 0.f ? v : 0.f;
}

// ---- global mean pool ----
__global__ void k_pool(const float* __restrict__ h, const int* __restrict__ batch,
                       float* __restrict__ gsum, float* __restrict__ gcnt) {
  int i = blockIdx.x * blockDim.x + threadIdx.x;
  if (i >= NN * HC) return;
  int n = i / HC, c = i % HC;
  int g = batch[n];
  atomicAdd(&gsum[(size_t)g * HC + c], h[i]);
  if (c == 0) atomicAdd(&gcnt[g], 1.0f);
}

__global__ void k_pool_div(float* __restrict__ gsum, const float* __restrict__ gcnt) {
  int i = blockIdx.x * blockDim.x + threadIdx.x;
  if (i >= GGR * HC) return;
  float c = gcnt[i / HC];
  c = c > 1.f ? c : 1.f;
  gsum[i] /= c;
}

// ---- final MLP head: [G,32] @ [32,1] + relu ----
__global__ void k_final(const float* __restrict__ g2, const float* __restrict__ lw3,
                        const float* __restrict__ lb3, float* __restrict__ outp) {
  int i = blockIdx.x * blockDim.x + threadIdx.x;
  if (i >= GGR) return;
  float s = lb3[0];
#pragma unroll
  for (int c = 0; c < HIDD; ++c) s += g2[i * HIDD + c] * lw3[c];
  outp[i] = s > 0.f ? s : 0.f;
}

extern "C" void kernel_launch(void* const* d_in, const int* in_sizes, int n_in,
                              void* d_out, int out_size, void* d_ws, size_t ws_size,
                              hipStream_t stream) {
  const float* x     = (const float*)d_in[0];
  const int*   ei    = (const int*)d_in[1];
  const float* ef    = (const float*)d_in[2];
  const int*   batch = (const int*)d_in[3];
  const float* W[3]  = {(const float*)d_in[5],  (const float*)d_in[11], (const float*)d_in[17]};
  const float* We[3] = {(const float*)d_in[6],  (const float*)d_in[12], (const float*)d_in[18]};
  const float* As[3] = {(const float*)d_in[7],  (const float*)d_in[13], (const float*)d_in[19]};
  const float* Ad[3] = {(const float*)d_in[8],  (const float*)d_in[14], (const float*)d_in[20]};
  const float* Ae[3] = {(const float*)d_in[9],  (const float*)d_in[15], (const float*)d_in[21]};
  const float* Bb[3] = {(const float*)d_in[10], (const float*)d_in[16], (const float*)d_in[22]};
  const float* lw1 = (const float*)d_in[23]; const float* lb1 = (const float*)d_in[24];
  const float* lw2 = (const float*)d_in[25]; const float* lb2 = (const float*)d_in[26];
  const float* lw3 = (const float*)d_in[27]; const float* lb3 = (const float*)d_in[28];
  const int* src0 = ei;
  const int* dst0 = ei + EE;

  char* wsp = (char*)d_ws;
  auto alloc = [&](size_t bytes) -> char* {
    char* p = wsp;
    wsp += (bytes + 255) & ~(size_t)255;
    return p;
  };
  float*    cnt       = (float*)alloc((size_t)NN * 4);
  float*    loop_attr = (float*)alloc((size_t)NN * FEDGE * 4);
  float*    hbuf      = (float*)alloc((size_t)NN * HC * 4);
  float*    xs        = (float*)alloc((size_t)NN * HC * 4);
  float*    outacc    = (float*)alloc((size_t)NN * HC * 4);
  float*    al_s      = (float*)alloc((size_t)NN * HEADS * 4);
  float*    al_d      = (float*)alloc((size_t)NN * HEADS * 4);
  float*    w_att     = (float*)alloc((size_t)FEDGE * HEADS * 4);
  float*    logits    = (float*)alloc((size_t)ETOT * HEADS * 4);
  unsigned* mord      = (unsigned*)alloc((size_t)NN * HEADS * 4);
  float*    denom     = (float*)alloc((size_t)NN * HEADS * 4);
  float*    gsum      = (float*)alloc((size_t)GGR * HC * 4);
  float*    gcnt      = (float*)alloc((size_t)GGR * 4);
  float*    g1        = (float*)alloc((size_t)GGR * HIDD * 4);
  float*    g2        = (float*)alloc((size_t)GGR * HIDD * 4);

  // self-loop edge attributes (PyG add_self_loops fill_value='mean')
  hipMemsetAsync(cnt, 0, (size_t)NN * 4, stream);
  hipMemsetAsync(loop_attr, 0, (size_t)NN * FEDGE * 4, stream);
  k_count_loop<<<(EE + 255) / 256, 256, 0, stream>>>(dst0, ef, cnt, loop_attr);
  k_loop_div<<<(NN * FEDGE + 255) / 256, 256, 0, stream>>>(loop_attr, cnt);

  const float* hin = x;
  int Kdim = FNODE;
  for (int l = 0; l < 3; ++l) {
    // xs = hin @ W : [N x K] @ [K x 128]; 4 waves/block cover all 128 cols.
    wmma_gemm<<<dim3(NN / 16, 1), 128, 0, stream>>>(hin, W[l], nullptr, xs,
                                                    NN, Kdim, HC, 0);
    k_node_att<<<(NN * HEADS + 255) / 256, 256, 0, stream>>>(xs, As[l], Ad[l], al_s, al_d);
    k_watt<<<1, 64, 0, stream>>>(We[l], Ae[l], w_att);
    k_init_m<<<(NN * HEADS + 255) / 256, 256, 0, stream>>>(mord, NN * HEADS);
    hipMemsetAsync(denom, 0, (size_t)NN * HEADS * 4, stream);
    hipMemsetAsync(outacc, 0, (size_t)NN * HC * 4, stream);
    k_logits<<<(ETOT + 255) / 256, 256, 0, stream>>>(src0, dst0, ef, loop_attr,
                                                     al_s, al_d, w_att, logits, mord);
    k_expsum<<<(ETOT + 255) / 256, 256, 0, stream>>>(dst0, logits, mord, denom);
    size_t aggN = (size_t)ETOT * HC;
    k_aggregate<<<(unsigned)((aggN + 255) / 256), 256, 0, stream>>>(src0, dst0, logits,
                                                                    denom, xs, outacc);
    k_bias_relu<<<(NN * HC + 255) / 256, 256, 0, stream>>>(outacc, Bb[l], hbuf);
    hin = hbuf;
    Kdim = HC;
  }

  // global mean pool + MLP head
  hipMemsetAsync(gsum, 0, (size_t)GGR * HC * 4, stream);
  hipMemsetAsync(gcnt, 0, (size_t)GGR * 4, stream);
  k_pool<<<(NN * HC + 255) / 256, 256, 0, stream>>>(hbuf, batch, gsum, gcnt);
  k_pool_div<<<(GGR * HC + 255) / 256, 256, 0, stream>>>(gsum, gcnt);
  wmma_gemm<<<dim3(GGR / 16, 1), 128, 0, stream>>>(gsum, lw1, lb1, g1, GGR, HC, HIDD, 1);
  wmma_gemm<<<dim3(GGR / 16, 1), 128, 0, stream>>>(g1, lw2, lb2, g2, GGR, HIDD, HIDD, 1);
  k_final<<<(GGR + 255) / 256, 256, 0, stream>>>(g2, lw3, lb3, (float*)d_out);
}
